// ZetaOrganism_55654186222144
// MI455X (gfx1250) — matrix operand order repaired and checked by reading, not compiled, not run on hardware
//
#include <hip/hip_runtime.h>
#include <hip/hip_bf16.h>
#include <math.h>

// ---------------------------------------------------------------------------
// ZetaOrganism step for MI455X (gfx1250, wave32, WMMA).
//
// Dominant work: pair = tanh(A @ S^T) with A = states@W_inf (8192x64),
// S = states, masked by neighbor graph, reduced to row/col means.
// Since nbr is symmetric, in_mean = row sums of masked tanh(S @ A^T),
// so each i-block computes both GEMM directions as row reductions only.
// All data L2-resident (~4MB << 192MB) -> compute bound -> f16 WMMA path
// (v_wmma_f32_16x16x32_f16, two chained K-halves per 16x16 tile).
// Epilogue tanh uses a branch-free exp/rcp identity (v_exp_f32 + v_rcp_f32)
// instead of libm tanhf's branchy expansion.
// ---------------------------------------------------------------------------

#define G 256
#define D 64
#define HD 128
#define NAGENT 8192

typedef __attribute__((ext_vector_type(16))) _Float16 v16h;
typedef __attribute__((ext_vector_type(8)))  _Float16 v8h;
typedef __attribute__((ext_vector_type(8)))  float    v8f;

union Frag { v16h v; v8h h2[2]; };

// ---------------- workspace layout (bytes) ----------------
#define WS_EGRID   0u
#define WS_RGRID   262144u
#define WS_TMP     524288u
#define WS_FIELD   786432u
#define WS_AH      1048576u   // 8192*64 f16
#define WS_SH      2097152u   // 8192*64 f16
#define WS_SUMOUT  3145728u   // 8192 f32
#define WS_SUMIN   3178496u
#define WS_NNBR    3211264u   // 8192 i32
#define WS_NMASS   3244032u
#define WS_NFI     3276800u
#define WS_FIKEY   3309568u

// tanh(x) = 1 - 2/(exp(2x)+1); exp->inf => +1, exp->0 => -1. Branch-free:
// v_mul, v_exp_f32, v_add, v_rcp_f32, v_fma. Fine for the masked pair sums.
__device__ __forceinline__ float fast_tanh(float x) {
    float e = __expf(2.0f * x);
    return 1.0f - 2.0f * __builtin_amdgcn_rcpf(e + 1.0f);
}

// ---------------- init: zero grids + reduction arrays, key = INT_MAX -------
__global__ void k_init(unsigned* __restrict__ grids, unsigned* __restrict__ sums,
                       int* __restrict__ key) {
    int t = blockIdx.x * 256 + threadIdx.x;
    if (t < 2 * G * G) grids[t] = 0u;                 // egrid + rgrid
    if (t < 5 * NAGENT) sums[t] = 0u;                 // sum_out..n_fi (contiguous)
    if (t < NAGENT)     key[t] = 0x7fffffff;
}

// ---------------- scatter-max energies / roles onto grid -------------------
__global__ void k_scatter(const int* __restrict__ pos, const float* __restrict__ ener,
                          const int* __restrict__ roles,
                          unsigned* __restrict__ egrid, unsigned* __restrict__ rgrid) {
    int i = blockIdx.x * 256 + threadIdx.x;
    if (i >= NAGENT) return;
    int x = pos[2 * i], y = pos[2 * i + 1];
    atomicMax(&egrid[y * G + x], __float_as_uint(ener[i]));         // >=0 floats:
    atomicMax(&rgrid[y * G + x], __float_as_uint((float)roles[i])); // bits monotone
}

// ---------------- prep: A = states@W_inf -> f16, S -> f16 ------------------
__global__ __launch_bounds__(256)
void k_prep(const float* __restrict__ states, const float* __restrict__ W_inf,
            _Float16* __restrict__ Ah, _Float16* __restrict__ Sh) {
    int t = blockIdx.x * 256 + threadIdx.x;   // t = i*64 + d, 524288 threads
    int i = t >> 6, d = t & 63;
    Sh[t] = (_Float16)states[t];
    float acc = 0.f;
    #pragma unroll 8
    for (int k = 0; k < 64; ++k) acc += states[i * 64 + k] * W_inf[k * 64 + d];
    Ah[t] = (_Float16)acc;
}

// ---------------- separable gaussian blur (M=15, sigma=2, zero pad) --------
__device__ __forceinline__ void gaussw(float* w) {
    float s = 0.f;
    #pragma unroll
    for (int t = 0; t < 15; ++t) { float x = (float)t - 7.0f; w[t] = expf(-(x * x) / 8.0f); s += w[t]; }
    #pragma unroll
    for (int t = 0; t < 15; ++t) w[t] /= s;
}

__global__ void k_blur_v(const float* __restrict__ eg, const float* __restrict__ rg,
                         float* __restrict__ tmp) {
    int p = blockIdx.x * 256 + threadIdx.x;
    int x = p & (G - 1), y = p >> 8;
    float w[15]; gaussw(w);
    float acc = 0.f;
    #pragma unroll
    for (int t = 0; t < 15; ++t) {
        int yy = y + t - 7;
        if (yy >= 0 && yy < G) acc += w[t] * (eg[yy * G + x] + 0.5f * rg[yy * G + x]);
    }
    tmp[p] = acc;
}

__global__ void k_blur_h(const float* __restrict__ tmp, float* __restrict__ field) {
    int p = blockIdx.x * 256 + threadIdx.x;
    int x = p & (G - 1), y = p >> 8;
    float w[15]; gaussw(w);
    float acc = 0.f;
    #pragma unroll
    for (int t = 0; t < 15; ++t) {
        int xx = x + t - 7;
        if (xx >= 0 && xx < G) acc += w[t] * tmp[y * G + xx];
    }
    field[p] = acc;
}

// ---------------- the big one: masked pairwise WMMA kernel -----------------
// gridDim = (256 i-blocks of 32 rows, 4 j-slices).
// block = 128 threads = 4 waves; wave (wi,wj) owns 16x16 subtile of 32x32.
__global__ __launch_bounds__(128)
void k_pair(const _Float16* __restrict__ Ah, const _Float16* __restrict__ Sh,
            const int* __restrict__ pos, const int* __restrict__ roles,
            float* __restrict__ sum_out, float* __restrict__ sum_in,
            int* __restrict__ n_nbr, int* __restrict__ n_mass,
            int* __restrict__ n_fi, int* __restrict__ fikey) {
    __shared__ float ls_out[32], ls_in[32];
    __shared__ unsigned long long ls_cnt[32];
    __shared__ int ls_key[32];

    const int lane = threadIdx.x & 31;
    const int w    = threadIdx.x >> 5;
    const int wi   = w >> 1, wj = w & 1;
    const int i0   = blockIdx.x * 32;
    const bool hi  = lane >= 16;
    const int ln   = lane & 15;

    if (threadIdx.x < 32) {
        ls_out[threadIdx.x] = 0.f; ls_in[threadIdx.x] = 0.f;
        ls_cnt[threadIdx.x] = 0ull; ls_key[threadIdx.x] = 0x7fffffff;
    }
    __syncthreads();

    // Hoisted A-operands (constant over j loop). CDNA5 16-bit A layout:
    // lane l -> row l&15; lo lanes K = {0..7,16..23}+32h, hi lanes +8.
    const int rowA = i0 + 16 * wi + ln;
    Frag aO[2], aI[2];
    #pragma unroll
    for (int h = 0; h < 2; ++h) {
        int kb = (hi ? 8 : 0) + 32 * h;
        aO[h].h2[0] = *(const v8h*)(Ah + rowA * 64 + kb);
        aO[h].h2[1] = *(const v8h*)(Ah + rowA * 64 + kb + 16);
        aI[h].h2[0] = *(const v8h*)(Sh + rowA * 64 + kb);
        aI[h].h2[1] = *(const v8h*)(Sh + rowA * 64 + kb + 16);
    }
    // Per-lane i rows of the C tile: lanes 0-15 rows r, lanes 16-31 rows 8+r.
    int ig8[8], xi8[8], yi8[8];
    #pragma unroll
    for (int r = 0; r < 8; ++r) {
        int ig = i0 + 16 * wi + (hi ? 8 + r : r);
        ig8[r] = ig; xi8[r] = pos[2 * ig]; yi8[r] = pos[2 * ig + 1];
    }

    const int jend = (blockIdx.y + 1) * (NAGENT / 4);
    for (int j0 = blockIdx.y * (NAGENT / 4); j0 < jend; j0 += 32) {
        // One fixed column j per lane across the whole subtile.
        const int jg = j0 + 16 * wj + ln;
        const int xj = pos[2 * jg], yj = pos[2 * jg + 1];
        const int rj = roles[jg];

        // Prefetch next chunk's B rows (global_prefetch_b8); address stays
        // inside the workspace allocation even on the last iteration.
        __builtin_prefetch(Sh + (jg + 32) * 64, 0, 1);
        __builtin_prefetch(Ah + (jg + 32) * 64, 0, 1);

        // B operands: lane l -> col l&15, K = {0..7,16..23}(+8 hi) +32h, matching A.
        Frag bO[2], bI[2];
        #pragma unroll
        for (int h = 0; h < 2; ++h) {
            int kb = (hi ? 8 : 0) + 32 * h;
            bO[h].h2[0] = *(const v8h*)(Sh + jg * 64 + kb);
            bO[h].h2[1] = *(const v8h*)(Sh + jg * 64 + kb + 16);
            bI[h].h2[0] = *(const v8h*)(Ah + jg * 64 + kb);
            bI[h].h2[1] = *(const v8h*)(Ah + jg * 64 + kb + 16);
        }

        v8f co = {}, ci = {};
        co = __builtin_amdgcn_wmma_f32_16x16x32_f16(false, aO[0].v, false, bO[0].v, (short)0, co, false, false);
        co = __builtin_amdgcn_wmma_f32_16x16x32_f16(false, aO[1].v, false, bO[1].v, (short)0, co, false, false);
        ci = __builtin_amdgcn_wmma_f32_16x16x32_f16(false, aI[0].v, false, bI[0].v, (short)0, ci, false, false);
        ci = __builtin_amdgcn_wmma_f32_16x16x32_f16(false, aI[1].v, false, bI[1].v, (short)0, ci, false, false);

        // Epilogue: mask + tanh + per-row reductions across 16-lane halves.
        #pragma unroll
        for (int r = 0; r < 8; ++r) {
            int dx = xi8[r] - xj, dy = yi8[r] - yj;
            int d2 = dx * dx + dy * dy;
            bool nbr = (d2 <= 25) && (ig8[r] != jg);
            float vo = nbr ? fast_tanh(co[r]) : 0.f;
            float vi = nbr ? fast_tanh(ci[r]) : 0.f;
            int pk  = nbr ? (1 | ((rj == 0) << 10) | ((rj == 1) << 20)) : 0;
            int key = (nbr && rj == 1) ? ((d2 << 13) | jg) : 0x7fffffff;
            #pragma unroll
            for (int m = 1; m < 16; m <<= 1) {
                vo += __shfl_xor(vo, m, 32);
                vi += __shfl_xor(vi, m, 32);
                pk += __shfl_xor(pk, m, 32);
                int ok = __shfl_xor(key, m, 32);
                key = ok < key ? ok : key;
            }
            if (ln == 0) {
                int row = 16 * wi + (hi ? 8 + r : r);
                atomicAdd(&ls_out[row], vo);
                atomicAdd(&ls_in[row], vi);
                unsigned long long wp = (unsigned long long)(pk & 1023)
                    | ((unsigned long long)((pk >> 10) & 1023) << 21)
                    | ((unsigned long long)((pk >> 20) & 1023) << 42);
                atomicAdd(&ls_cnt[row], wp);
                atomicMin(&ls_key[row], key);
            }
        }
    }
    __syncthreads();
    if (threadIdx.x < 32) {
        int ig = i0 + threadIdx.x;
        unsigned long long c = ls_cnt[threadIdx.x];
        atomicAdd(&sum_out[ig], ls_out[threadIdx.x]);
        atomicAdd(&sum_in[ig],  ls_in[threadIdx.x]);
        atomicAdd(&n_nbr[ig],  (int)(c & 0x1fffff));
        atomicAdd(&n_mass[ig], (int)((c >> 21) & 0x1fffff));
        atomicAdd(&n_fi[ig],   (int)((c >> 42) & 0x1fffff));
        atomicMin(&fikey[ig],  ls_key[threadIdx.x]);
    }
}

// ---------------- finalize: MLP, energy, roles, movement -------------------
__global__ __launch_bounds__(256)
void k_final(const int* __restrict__ pos, const float* __restrict__ states,
             const int* __restrict__ roles, const float* __restrict__ ener,
             const float* __restrict__ W_self, const float* __restrict__ W1,
             const float* __restrict__ b1, const float* __restrict__ W2,
             const float* __restrict__ b2, const float* __restrict__ field,
             const float* __restrict__ sum_out, const float* __restrict__ sum_in,
             const int* __restrict__ n_nbr, const int* __restrict__ n_mass,
             const int* __restrict__ n_fi, const int* __restrict__ fikey,
             float* __restrict__ out_states, float* __restrict__ out_e,
             int* __restrict__ out_roles, int* __restrict__ out_pos) {
    int i = blockIdx.x * 256 + threadIdx.x;
    if (i >= NAGENT) return;
    int x = pos[2 * i], y = pos[2 * i + 1];

    float f00 = field[y * G + x];
    float pot = f00;
    float gx, gy;                     // jnp.gradient: central interior, 1-sided edges
    if (x == 0)          gx = field[y * G + 1] - f00;
    else if (x == G - 1) gx = f00 - field[y * G + G - 2];
    else                 gx = 0.5f * (field[y * G + x + 1] - field[y * G + x - 1]);
    if (y == 0)          gy = field[G + x] - f00;
    else if (y == G - 1) gy = f00 - field[(G - 2) * G + x];
    else                 gy = 0.5f * (field[(y + 1) * G + x] - field[(y - 1) * G + x]);

    int n = n_nbr[i], mass = n_mass[i], fi = n_fi[i], key = fikey[i];
    bool has = n > 0;
    float denom = (float)max(n, 1);
    float net = has ? (sum_out[i] - sum_in[i]) / denom : 0.f;

    float s[64];
    #pragma unroll
    for (int d = 0; d < 64; ++d) s[d] = states[i * 64 + d];

    float v[65];
    for (int d = 0; d < 64; ++d) {
        float acc = 0.f;
        #pragma unroll 8
        for (int k = 0; k < 64; ++k) acc += s[k] * W_self[k * 64 + d];
        v[d] = s[d] + 0.1f * tanhf(acc);
    }
    v[64] = pot;

    float acc2[64];
    #pragma unroll
    for (int d = 0; d < 64; ++d) acc2[d] = b2[d];
    for (int j = 0; j < HD; ++j) {
        float hj = b1[j];
        #pragma unroll 5
        for (int k = 0; k < 65; ++k) hj += v[k] * W1[k * HD + j];
        hj = tanhf(hj);
        #pragma unroll 8
        for (int d = 0; d < 64; ++d) acc2[d] += hj * W2[j * 64 + d];
    }
    for (int d = 0; d < 64; ++d)
        out_states[i * 64 + d] = has ? (acc2[d] + 0.3f * s[d]) : s[d];

    float e = ener[i];
    int role = roles[i];
    float e_fi  = e + 0.02f * (float)mass;
    float e_oth = e * 0.995f + 0.05f * fmaxf(pot, 0.f) + (mass >= 3 ? 0.02f : 0.f);
    float ne = (role == 1 ? e_fi : e_oth) + 0.02f * fmaxf(net, 0.f);
    ne = fminf(fmaxf(ne, 0.f), 1.f);
    float score = net + 0.5f;
    int r0 = ((ne > 0.7f) && (mass >= 2) && ((fi == 0) || (score > 0.2f))) ? 1 : 0;
    int r1 = ((mass < 1) || (ne < 0.2f) || (score < -0.3f)) ? 0 : 1;
    int r2 = ((fi > 0) && (net > 0.5f)) ? 1 : ((ne < 0.2f) ? 0 : 2);
    int nr = (role == 0) ? r0 : ((role == 1) ? r1 : r2);

    bool move_mass = (nr == 0) && ((fabsf(gx) > 0.01f) || (fabsf(gy) > 0.01f));
    bool move_c    = (nr == 2) && (fi > 0);
    int dx = 0, dy = 0;
    if (move_mass) {
        dx = (gx > 0.f) - (gx < 0.f);
        dy = (gy > 0.f) - (gy < 0.f);
    } else if (move_c) {
        int tgt = (key == 0x7fffffff) ? 0 : (key & 8191);
        int xt = pos[2 * tgt], yt = pos[2 * tgt + 1];
        dx = (xt > x) - (xt < x);
        dy = (yt > y) - (yt < y);
    }
    out_e[i] = ne;
    out_roles[i] = nr;
    out_pos[2 * i]     = min(max(x + dx, 0), G - 1);
    out_pos[2 * i + 1] = min(max(y + dy, 0), G - 1);
}

// ---------------------------------------------------------------------------
extern "C" void kernel_launch(void* const* d_in, const int* in_sizes, int n_in,
                              void* d_out, int out_size, void* d_ws, size_t ws_size,
                              hipStream_t stream) {
    const int*   positions = (const int*)d_in[0];
    const float* states    = (const float*)d_in[1];
    const int*   roles     = (const int*)d_in[2];
    const float* energies  = (const float*)d_in[3];
    const float* W_inf     = (const float*)d_in[4];
    const float* W_self    = (const float*)d_in[5];
    const float* W1        = (const float*)d_in[6];
    const float* b1        = (const float*)d_in[7];
    const float* W2        = (const float*)d_in[8];
    const float* b2        = (const float*)d_in[9];

    char* ws = (char*)d_ws;
    float*     egrid   = (float*)(ws + WS_EGRID);
    float*     rgrid   = (float*)(ws + WS_RGRID);
    float*     tmp     = (float*)(ws + WS_TMP);
    float*     field   = (float*)(ws + WS_FIELD);
    _Float16*  Ah      = (_Float16*)(ws + WS_AH);
    _Float16*  Sh      = (_Float16*)(ws + WS_SH);
    float*     sum_out = (float*)(ws + WS_SUMOUT);
    float*     sum_in  = (float*)(ws + WS_SUMIN);
    int*       nnbr    = (int*)(ws + WS_NNBR);
    int*       nmass   = (int*)(ws + WS_NMASS);
    int*       nfi     = (int*)(ws + WS_NFI);
    int*       fikey   = (int*)(ws + WS_FIKEY);

    float* out_states = (float*)d_out;                    // 8192*64
    float* out_e      = out_states + NAGENT * 64;         // 8192
    int*   out_roles  = (int*)(out_e + NAGENT);           // 8192
    int*   out_pos    = out_roles + NAGENT;               // 8192*2

    k_init<<<(2 * G * G + 255) / 256, 256, 0, stream>>>((unsigned*)egrid,
                                                        (unsigned*)sum_out, fikey);
    k_scatter<<<(NAGENT + 255) / 256, 256, 0, stream>>>(positions, energies, roles,
                                                        (unsigned*)egrid, (unsigned*)rgrid);
    k_prep<<<(NAGENT * 64) / 256, 256, 0, stream>>>(states, W_inf, Ah, Sh);
    k_blur_v<<<(G * G) / 256, 256, 0, stream>>>(egrid, rgrid, tmp);
    k_blur_h<<<(G * G) / 256, 256, 0, stream>>>(tmp, field);
    k_pair<<<dim3(NAGENT / 32, 4), 128, 0, stream>>>(Ah, Sh, positions, roles,
                                                     sum_out, sum_in, nnbr, nmass,
                                                     nfi, fikey);
    k_final<<<(NAGENT + 255) / 256, 256, 0, stream>>>(positions, states, roles, energies,
                                                      W_self, W1, b1, W2, b2, field,
                                                      sum_out, sum_in, nnbr, nmass, nfi,
                                                      fikey, out_states, out_e,
                                                      out_roles, out_pos);
}
